// Model_79517024518378
// MI455X (gfx1250) — compile-verified, matrix operand order checked by reading, not run on hardware
//
#include <hip/hip_runtime.h>

#define Bdim 4096
#define Vdim 4096
#define Hdim 4096
#define Kdim (Vdim + Hdim)   // 8192

typedef __attribute__((ext_vector_type(16))) __bf16 v16bf;
typedef __attribute__((ext_vector_type(8)))  float  v8f;

// TDM descriptor groups (clang-23 amdgpu-toolchain 6-arg tensor_load_to_lds:
//   (uint32x4 g0, int32x8 g1, int32x4, int32x4, int32x8, i32 cpol))
typedef __attribute__((ext_vector_type(4))) unsigned tdm_g0_t;
typedef __attribute__((ext_vector_type(8))) int      tdm_g1_t;
typedef __attribute__((ext_vector_type(4))) int      tdm_g2_t;

union FragBF { unsigned u[8]; v16bf v; };

// Native f32->bf16 (RNE) via hardware cvt.
__device__ __forceinline__ unsigned short f2bfu(float f) {
  return __builtin_bit_cast(unsigned short, (__bf16)f);
}
__device__ __forceinline__ unsigned pack2(float a, float b) {
  return (unsigned)f2bfu(a) | ((unsigned)f2bfu(b) << 16);
}
__device__ __forceinline__ float sigm(float x) {
  return 1.0f / (1.0f + __expf(-x));
}
__device__ __forceinline__ float tanh_fast(float x) {
  x = fminf(fmaxf(x, -15.0f), 15.0f);   // keep __expf finite
  float e = __expf(2.0f * x);
  return (e - 1.0f) / (e + 1.0f);
}

// ---------------------------------------------------------------------------
// TDM: async 2D tile load (bf16) global -> LDS with 16B pad per 64B row.
// tile = BM rows x 32 bf16 cols; LDS pitch = 80B (40 ushorts) = our LP.
// ---------------------------------------------------------------------------
#define BM 128
#define BN 64
#define BK 32
#define LP 40   // LDS pitch in ushorts: 64B data + 16B pad per row

__device__ __forceinline__ void tdm_load_a(const unsigned short* gsrc,
                                           unsigned ldsOff,
                                           unsigned tensorD0 /* row len elems */) {
  unsigned long long ga = (unsigned long long)(uintptr_t)gsrc;
  tdm_g0_t g0;
  g0[0] = 1u;                                               // count=1, user D#
  g0[1] = ldsOff;                                           // lds_addr (bytes)
  g0[2] = (unsigned)ga;                                     // global_addr[31:0]
  g0[3] = (unsigned)((ga >> 32) & 0x1FFFFFFu) | (2u << 30); // addr[56:32]|type=2

  tdm_g1_t g1 = {};
  g1[0] = (int)(0x00010000u        // data_size = 1 (2 bytes), no multicast
        | (1u << 20)               // pad_enable
        | (3u << 22)               // pad_interval: 16 DWORDs (64B)
        | (3u << 25));             // pad_amount:   4 DWORDs (16B)
  g1[1] = (int)((tensorD0 & 0xFFFFu) << 16);                       // tensor_dim0 lo
  g1[2] = (int)(((tensorD0 >> 16) & 0xFFFFu) |
                ((unsigned)(Bdim & 0xFFFF) << 16));                // dim0 hi | dim1 lo
  g1[3] = (int)(((unsigned)Bdim >> 16) | ((unsigned)BK << 16));    // dim1 hi | tile_dim0
  g1[4] = (int)(BM);                                               // tile_dim1 | tile_dim2=0
  g1[5] = (int)tensorD0;                                           // tensor_dim0_stride lo
  g1[6] = 0;                                                       // stride hi | dim1_stride
  g1[7] = 0;

  __builtin_amdgcn_tensor_load_to_lds(g0, g1, (tdm_g2_t)0, (tdm_g2_t)0,
                                      (tdm_g1_t)0, 0);
}

// W staging body: 32 fp32 -> 16 packed bf16 uints into LDS.  Inlined per-gate
// so the load base is directly a kernel arg => global_load (not flat).
__device__ __forceinline__ void stage_w32(const float* __restrict__ src,
                                          unsigned* __restrict__ dst) {
  __builtin_prefetch(src + BK, 0, 3);   // next K-slice (global_prefetch_b8)
#pragma unroll
  for (int q = 0; q < 8; ++q) {
    float4 w = *(const float4*)(src + q * 4);
    dst[2*q]   = pack2(w.x, w.y);
    dst[2*q+1] = pack2(w.z, w.w);
  }
}

// ---------------------------------------------------------------------------
// Kernel 0: pack xh = cat(x, h) -> bf16 [B][K].  64 MB -> L2-resident for GEMM.
// ---------------------------------------------------------------------------
__device__ __forceinline__ void pack8(const float* __restrict__ src,
                                      unsigned short* __restrict__ dst) {
  float4 p0 = *(const float4*)src;
  float4 p1 = *(const float4*)(src + 4);
  uint4 o;
  o.x = pack2(p0.x, p0.y); o.y = pack2(p0.z, p0.w);
  o.z = pack2(p1.x, p1.y); o.w = pack2(p1.z, p1.w);
  *(uint4*)dst = o;
}

__global__ __launch_bounds__(256) void pack_xh(const float* __restrict__ x,
                                               const float* __restrict__ h,
                                               unsigned short* __restrict__ xh) {
  size_t e = ((size_t)blockIdx.x * 256 + threadIdx.x) * 8;  // 8 elems/thread
  int b = (int)(e / Kdim);
  int k = (int)(e % Kdim);  // chunks of 8 never straddle V boundary (V%8==0)
  if (k < Vdim) pack8(x + (size_t)b * Vdim + k,          xh + e);
  else          pack8(h + (size_t)b * Hdim + (k - Vdim), xh + e);
}

// ---------------------------------------------------------------------------
// Kernel 1: fused 4-gate GEMM + LSTM pointwise.  Tile 128x64, BK=32, 8 waves.
// A tile staged by the TDM (async tensor load); W converted fp32->bf16 in VALU.
// ---------------------------------------------------------------------------
__global__ __launch_bounds__(256) void lstm_gates(
    const unsigned short* __restrict__ xh,   // [B][K] bf16
    const float* __restrict__ cIn,           // [B][H]
    const float* __restrict__ Wf, const float* __restrict__ Wi,
    const float* __restrict__ Wc, const float* __restrict__ Wo,
    const float* __restrict__ bf, const float* __restrict__ bi,
    const float* __restrict__ bc, const float* __restrict__ bo,
    unsigned short* __restrict__ hOut)       // [B][H] bf16
{
  __shared__ alignas(16) unsigned short ldsA[BM * LP];
  __shared__ alignas(16) unsigned short ldsW[4][BN * LP];

  const int t    = threadIdx.x;
  const int lane = t & 31;
  const int wave = t >> 5;
  const int wm   = wave >> 1;   // 0..3  -> rows wm*32 .. +31
  const int wn   = wave & 1;    // 0..1  -> cols wn*32 .. +31
  const int b0   = blockIdx.x * BM;
  const int n0   = blockIdx.y * BN;

  const unsigned ldsABase = (unsigned)(uintptr_t)(void*)&ldsA[0];

  v8f acc[4][2][2] = {};

  // W staging: one 32-float row of one gate per thread; gate = wave>>1 (uniform)
  const int gW = t >> 6;            // 0..3  (wave-uniform)
  const int nW = t & 63;            // 0..63

  const int KT = Kdim / BK;         // 256
  for (int kt = 0; kt < KT; ++kt) {
    const int k0 = kt * BK;
    __syncthreads();
    if (wave == 0) {  // one TDM op stages the whole 128x32 bf16 A tile
      tdm_load_a(xh + (size_t)b0 * Kdim + k0, ldsABase, Kdim);
    }
    { // stage W: fp32 -> bf16; uniform branch keeps each base a kernel arg
      const size_t wOff = (size_t)(n0 + nW) * Kdim + k0;
      unsigned* dst = (unsigned*)&ldsW[gW][nW * LP];
      if      (gW == 0) stage_w32(Wf + wOff, dst);
      else if (gW == 1) stage_w32(Wi + wOff, dst);
      else if (gW == 2) stage_w32(Wc + wOff, dst);
      else              stage_w32(Wo + wOff, dst);
    }
    if (wave == 0) __builtin_amdgcn_s_wait_tensorcnt(0);
    __syncthreads();

    // 16-bit A/B fragment K-pair offsets per ISA layout (lanes 16..31: +8)
    const int half = lane >> 4;
    int kj[8];
#pragma unroll
    for (int j = 0; j < 8; ++j)
      kj[j] = ((j < 4) ? (2 * j) : (16 + 2 * (j - 4))) + 8 * half;

    FragBF fa[2];
#pragma unroll
    for (int i = 0; i < 2; ++i) {
      const int m = wm * 32 + i * 16 + (lane & 15);
#pragma unroll
      for (int j = 0; j < 8; ++j)
        fa[i].u[j] = *(const unsigned*)&ldsA[m * LP + kj[j]];
    }
#pragma unroll
    for (int g = 0; g < 4; ++g) {
#pragma unroll
      for (int j2 = 0; j2 < 2; ++j2) {
        FragBF fb;
        const int n = wn * 32 + j2 * 16 + (lane & 15);
#pragma unroll
        for (int j = 0; j < 8; ++j)
          fb.u[j] = *(const unsigned*)&ldsW[g][n * LP + kj[j]];
#pragma unroll
        for (int i = 0; i < 2; ++i)
          acc[g][i][j2] = __builtin_amdgcn_wmma_f32_16x16x32_bf16(
              false, fa[i].v, false, fb.v, (short)0, acc[g][i][j2],
              false, false);
      }
    }
  }

  // Epilogue: biases + sigmoid/tanh + c/h update, write h_new as bf16.
  // C/D layout: VGPR r, lanes 0-15 -> M=r, lanes 16-31 -> M=r+8; N=lane&15.
#pragma unroll
  for (int i = 0; i < 2; ++i) {
#pragma unroll
    for (int j2 = 0; j2 < 2; ++j2) {
      const int nIdx  = n0 + wn * 32 + j2 * 16 + (lane & 15);
      const int bBase = b0 + wm * 32 + i * 16 + 8 * (lane >> 4);
      const float vbf = bf[nIdx], vbi = bi[nIdx];
      const float vbc = bc[nIdx], vbo = bo[nIdx];
#pragma unroll
      for (int r = 0; r < 8; ++r) {
        const int bIdx = bBase + r;
        float fg = sigm(acc[0][i][j2][r] + vbf);
        float ig = sigm(acc[1][i][j2][r] + vbi);
        float cc = tanh_fast(acc[2][i][j2][r] + vbc);
        float og = sigm(acc[3][i][j2][r] + vbo);
        float cn = cIn[(size_t)bIdx * Hdim + nIdx] * fg + cc * ig;
        hOut[(size_t)bIdx * Hdim + nIdx] = f2bfu(tanh_fast(cn) * og);
      }
    }
  }
}

// ---------------------------------------------------------------------------
// Kernel 2: logits = h_new @ Wout^T + bout  -> d_out.  Same tiling, 1 matrix.
// ---------------------------------------------------------------------------
__global__ __launch_bounds__(256) void out_gemm(
    const unsigned short* __restrict__ hN,   // [B][H] bf16
    const float* __restrict__ Wout,          // [V][H]
    const float* __restrict__ bout,          // [V]
    float* __restrict__ logits)              // [B][V]
{
  __shared__ alignas(16) unsigned short ldsA[BM * LP];
  __shared__ alignas(16) unsigned short ldsW[BN * LP];

  const int t    = threadIdx.x;
  const int lane = t & 31;
  const int wave = t >> 5;
  const int wm   = wave >> 1;
  const int wn   = wave & 1;
  const int b0   = blockIdx.x * BM;
  const int n0   = blockIdx.y * BN;

  const unsigned ldsABase = (unsigned)(uintptr_t)(void*)&ldsA[0];

  v8f acc[2][2] = {};

  const int nW = t >> 2;            // 0..63
  const int kW = (t & 3) * 8;       // 0,8,16,24 (float units)

  const int KT = Hdim / BK;         // 128
  for (int kt = 0; kt < KT; ++kt) {
    const int k0 = kt * BK;
    __syncthreads();
    if (wave == 0) {
      tdm_load_a(hN + (size_t)b0 * Hdim + k0, ldsABase, Hdim);
    }
    {
      const float* src = Wout + (size_t)(n0 + nW) * Hdim + k0 + kW;
      __builtin_prefetch(src + BK, 0, 3);
      float4 w0 = *(const float4*)src;
      float4 w1 = *(const float4*)(src + 4);
      unsigned* dst = (unsigned*)&ldsW[nW * LP + kW];
      dst[0] = pack2(w0.x, w0.y); dst[1] = pack2(w0.z, w0.w);
      dst[2] = pack2(w1.x, w1.y); dst[3] = pack2(w1.z, w1.w);
    }
    if (wave == 0) __builtin_amdgcn_s_wait_tensorcnt(0);
    __syncthreads();

    const int half = lane >> 4;
    int kj[8];
#pragma unroll
    for (int j = 0; j < 8; ++j)
      kj[j] = ((j < 4) ? (2 * j) : (16 + 2 * (j - 4))) + 8 * half;

    FragBF fa[2];
#pragma unroll
    for (int i = 0; i < 2; ++i) {
      const int m = wm * 32 + i * 16 + (lane & 15);
#pragma unroll
      for (int j = 0; j < 8; ++j)
        fa[i].u[j] = *(const unsigned*)&ldsA[m * LP + kj[j]];
    }
#pragma unroll
    for (int j2 = 0; j2 < 2; ++j2) {
      FragBF fb;
      const int n = wn * 32 + j2 * 16 + (lane & 15);
#pragma unroll
      for (int j = 0; j < 8; ++j)
        fb.u[j] = *(const unsigned*)&ldsW[n * LP + kj[j]];
#pragma unroll
      for (int i = 0; i < 2; ++i)
        acc[i][j2] = __builtin_amdgcn_wmma_f32_16x16x32_bf16(
            false, fa[i].v, false, fb.v, (short)0, acc[i][j2], false, false);
    }
  }

#pragma unroll
  for (int i = 0; i < 2; ++i) {
#pragma unroll
    for (int j2 = 0; j2 < 2; ++j2) {
      const int nIdx  = n0 + wn * 32 + j2 * 16 + (lane & 15);
      const int bBase = b0 + wm * 32 + i * 16 + 8 * (lane >> 4);
      const float bb = bout[nIdx];
#pragma unroll
      for (int r = 0; r < 8; ++r)
        logits[(size_t)(bBase + r) * Vdim + nIdx] = acc[i][j2][r] + bb;
    }
  }
}

// ---------------------------------------------------------------------------
// Kernel 3: in-place row softmax over V=4096 (one block per row; values
// cached in registers between the max/sum passes -> single HBM round trip).
// ---------------------------------------------------------------------------
__global__ __launch_bounds__(256) void softmax_rows(float* __restrict__ out) {
  __shared__ float red[256];
  const int row = blockIdx.x;
  const int t = threadIdx.x;
  float v[16];
  float mx = -3.4e38f;
#pragma unroll
  for (int i = 0; i < 16; ++i) {
    v[i] = out[(size_t)row * Vdim + t + i * 256];
    mx = fmaxf(mx, v[i]);
  }
  red[t] = mx; __syncthreads();
  for (int s = 128; s > 0; s >>= 1) {
    if (t < s) red[t] = fmaxf(red[t], red[t + s]);
    __syncthreads();
  }
  mx = red[0]; __syncthreads();
  float sum = 0.f;
#pragma unroll
  for (int i = 0; i < 16; ++i) { v[i] = __expf(v[i] - mx); sum += v[i]; }
  red[t] = sum; __syncthreads();
  for (int s = 128; s > 0; s >>= 1) {
    if (t < s) red[t] += red[t + s];
    __syncthreads();
  }
  const float inv = 1.0f / red[0];
#pragma unroll
  for (int i = 0; i < 16; ++i)
    out[(size_t)row * Vdim + t + i * 256] = v[i] * inv;
}

// ---------------------------------------------------------------------------
extern "C" void kernel_launch(void* const* d_in, const int* in_sizes, int n_in,
                              void* d_out, int out_size, void* d_ws, size_t ws_size,
                              hipStream_t stream) {
  const float* x    = (const float*)d_in[0];
  const float* h    = (const float*)d_in[1];
  const float* c    = (const float*)d_in[2];
  const float* Wf   = (const float*)d_in[3];
  const float* bf   = (const float*)d_in[4];
  const float* Wi   = (const float*)d_in[5];
  const float* bi   = (const float*)d_in[6];
  const float* Wc   = (const float*)d_in[7];
  const float* bc   = (const float*)d_in[8];
  const float* Wo   = (const float*)d_in[9];
  const float* bo   = (const float*)d_in[10];
  const float* Wout = (const float*)d_in[11];
  const float* bout = (const float*)d_in[12];
  float* out = (float*)d_out;

  // ws layout: [0,64MB) xh bf16 [B][K]; [64MB,96MB) h_new bf16 [B][H]
  unsigned short* xh_bf = (unsigned short*)d_ws;
  unsigned short* hn_bf = xh_bf + (size_t)Bdim * Kdim;

  const int packBlocks = (int)(((size_t)Bdim * Kdim) / (256 * 8));  // 16384
  pack_xh<<<packBlocks, 256, 0, stream>>>(x, h, xh_bf);
  lstm_gates<<<dim3(Bdim / BM, Hdim / BN), 256, 0, stream>>>(
      xh_bf, c, Wf, Wi, Wc, Wo, bf, bi, bc, bo, hn_bf);
  out_gemm<<<dim3(Bdim / BM, Vdim / BN), 256, 0, stream>>>(
      hn_bf, Wout, bout, out);
  softmax_rows<<<Bdim, 256, 0, stream>>>(out);
}
